// Model_15135464751445
// MI455X (gfx1250) — matrix operand order repaired and checked by reading, not compiled, not run on hardware
//
#include <hip/hip_runtime.h>
#include <math.h>

#define B_ 4
#define L_ 96
#define N_ 200
#define D_ 128
#define H_ 8
#define DH_ 16
#define DFF_ 512
#define E_ 4
#define LAYERS_ 3
#define PRED_ 96
#define TODN_ 288
#define TOK_ (B_*N_*L_)   /* 76800 tokens of main stack, layout [B,N,L,D] */
#define SEQ_ (B_*N_)      /* 800 sequences */
#define BLN_ (B_*L_*N_)   /* 76800 */
#define EPS_ 1e-5f

typedef __attribute__((ext_vector_type(16))) _Float16 v16h;
typedef __attribute__((ext_vector_type(8)))  float    v8f;

#define GF_ACC  1
#define GF_RELU 2

// ---------------------------------------------------------------------------
// WMMA GEMM: C[M,N] (=|+=) optional_rowscale * (relu?)(A[M,K] @ B[K,N] + bias)
// fp32 in memory, f16 WMMA fragments, fp32 accumulation.
// Block = 256 threads (8 waves) computing a 64x64 tile; per 32-K step the
// block stages A(64x32) and B(32x64) into LDS as f16 via float4 global loads.
// Each wave owns a 32Mx16N strip = two v_wmma_f32_16x16x32_f16 accumulators.
// Requires K % 32 == 0 (true for all uses here: 96,128,12288).
// ---------------------------------------------------------------------------
__global__ __launch_bounds__(256)
void gemm_ws_kernel(const float* __restrict__ A, int lda,
                    const float* __restrict__ Bm, int ldb,
                    const float* __restrict__ bias,
                    const float* __restrict__ rowscale, int rss,
                    float* __restrict__ C, int ldc,
                    int M, int K, int Ncols, int flags)
{
    __shared__ _Float16 lA[64 * 32];   // [m local 64][k local 32]
    __shared__ _Float16 lB[32 * 64];   // [k local 32][n local 64]

    int tid  = threadIdx.x;
    int wave = tid >> 5;
    int lane = tid & 31;
    int m0 = blockIdx.y * 64;
    int n0 = blockIdx.x * 64;
    int mr = lane & 15;
    int hh = lane >> 4;

    int mbase = (wave >> 2) * 32;      // 0 or 32 within M tile
    int nbase = (wave & 3) * 16;       // 0/16/32/48 within N tile

    // cooperative staging coordinates (8 contiguous floats per thread)
    int ar  = (tid * 8) >> 5;          // A local row 0..63
    int ak  = (tid * 8) & 31;          // A local k   0..24 step 8
    int bkr = (tid * 8) >> 6;          // B local k   0..31
    int bnn = (tid * 8) & 63;          // B local n   0..56 step 8

    v8f acc0 = {0.f,0.f,0.f,0.f,0.f,0.f,0.f,0.f};
    v8f acc1 = {0.f,0.f,0.f,0.f,0.f,0.f,0.f,0.f};

    for (int k0 = 0; k0 < K; k0 += 32) {
        __syncthreads();
        // ---- stage A tile ----
        {
            int grow = m0 + ar;
            _Float16* dst = &lA[ar * 32 + ak];
            if (grow < M) {
                const float* src = A + (size_t)grow * lda + k0 + ak;
                float4 f0 = *(const float4*)src;
                float4 f1 = *(const float4*)(src + 4);
                dst[0] = (_Float16)f0.x; dst[1] = (_Float16)f0.y;
                dst[2] = (_Float16)f0.z; dst[3] = (_Float16)f0.w;
                dst[4] = (_Float16)f1.x; dst[5] = (_Float16)f1.y;
                dst[6] = (_Float16)f1.z; dst[7] = (_Float16)f1.w;
            } else {
#pragma unroll
                for (int j = 0; j < 8; ++j) dst[j] = (_Float16)0.f;
            }
        }
        // ---- stage B tile ----
        {
            int gcol = n0 + bnn;
            _Float16* dst = &lB[bkr * 64 + bnn];
            const float* src = Bm + (size_t)(k0 + bkr) * ldb + gcol;
            if (gcol + 7 < Ncols) {
                float4 f0 = *(const float4*)src;
                float4 f1 = *(const float4*)(src + 4);
                dst[0] = (_Float16)f0.x; dst[1] = (_Float16)f0.y;
                dst[2] = (_Float16)f0.z; dst[3] = (_Float16)f0.w;
                dst[4] = (_Float16)f1.x; dst[5] = (_Float16)f1.y;
                dst[6] = (_Float16)f1.z; dst[7] = (_Float16)f1.w;
            } else {
#pragma unroll
                for (int j = 0; j < 8; ++j)
                    dst[j] = (gcol + j < Ncols) ? (_Float16)src[j] : (_Float16)0.f;
            }
        }
        __syncthreads();

        // ---- build fragments from LDS ----
        v16h bf;
#pragma unroll
        for (int j = 0; j < 8; ++j) {              // B: col per lane, interleaved K
            int kb = ((j & 4) << 2) + hh * 8 + ((j & 3) << 1);
            bf[2*j]   = lB[kb * 64 + nbase + mr];
            bf[2*j+1] = lB[(kb + 1) * 64 + nbase + mr];
        }
        v16h af0, af1;
#pragma unroll
        for (int j = 0; j < 8; ++j) {              // A: row per lane, interleaved K
            int ka = ((j & 4) << 2) + hh * 8 + ((j & 3) << 1);
            const _Float16* pa0 = &lA[(mbase + mr) * 32 + ka];
            const _Float16* pa1 = &lA[(mbase + 16 + mr) * 32 + ka];
            af0[2*j] = pa0[0]; af0[2*j+1] = pa0[1];
            af1[2*j] = pa1[0]; af1[2*j+1] = pa1[1];
        }
        acc0 = __builtin_amdgcn_wmma_f32_16x16x32_f16(false, af0, false, bf,
                                                      (short)0, acc0, false, false);
        acc1 = __builtin_amdgcn_wmma_f32_16x16x32_f16(false, af1, false, bf,
                                                      (short)0, acc1, false, false);
    }

    // ---- epilogue ----
    int ncol = n0 + nbase + mr;
    if (ncol >= Ncols) return;
    float bv = bias ? bias[ncol] : 0.f;
#pragma unroll
    for (int r = 0; r < 8; ++r) {
        int row0 = m0 + mbase + r + hh * 8;
        int row1 = row0 + 16;
        if (row0 < M) {
            float v = acc0[r] + bv;
            if (flags & GF_RELU) v = fmaxf(v, 0.f);
            if (rowscale) v *= rowscale[(size_t)row0 * rss];
            float* cp = C + (size_t)row0 * ldc + ncol;
            if (flags & GF_ACC) *cp += v; else *cp = v;
        }
        if (row1 < M) {
            float v = acc1[r] + bv;
            if (flags & GF_RELU) v = fmaxf(v, 0.f);
            if (rowscale) v *= rowscale[(size_t)row1 * rss];
            float* cp = C + (size_t)row1 * ldc + ncol;
            if (flags & GF_ACC) *cp += v; else *cp = v;
        }
    }
}

// ---------------------------------------------------------------------------
// RevIN statistics: per-batch mean/std over L*N elements
// ---------------------------------------------------------------------------
__global__ void revin_stats_kernel(const float* __restrict__ x,
                                   float* __restrict__ mean, float* __restrict__ stdv)
{
    int b = blockIdx.x;
    __shared__ float s1[256], s2[256];
    float a = 0.f, q = 0.f;
    for (int i = threadIdx.x; i < L_*N_; i += 256) {
        float v = x[(size_t)b * L_ * N_ + i];
        a += v; q += v * v;
    }
    s1[threadIdx.x] = a; s2[threadIdx.x] = q; __syncthreads();
    for (int s = 128; s > 0; s >>= 1) {
        if (threadIdx.x < s) { s1[threadIdx.x] += s1[threadIdx.x+s]; s2[threadIdx.x] += s2[threadIdx.x+s]; }
        __syncthreads();
    }
    if (threadIdx.x == 0) {
        float m = s1[0] / (float)(L_*N_);
        float v = s2[0] / (float)(L_*N_) - m * m;
        mean[b] = m; stdv[b] = sqrtf(v + EPS_);
    }
}

// xn / trend / season / spatial-input (xn transposed to [B,N,L])
__global__ void prep_kernel(const float* __restrict__ x, const float* __restrict__ mean,
                            const float* __restrict__ stdv, float* __restrict__ trend,
                            float* __restrict__ season, float* __restrict__ spin)
{
    int idx = blockIdx.x * blockDim.x + threadIdx.x;
    if (idx >= BLN_) return;
    int n = idx % N_; int l = (idx / N_) % L_; int b = idx / (N_ * L_);
    float m = mean[b], s = stdv[b];
    float xc = (x[idx] - m) / s;
    float xm = (l > 0)       ? (x[idx - N_] - m) / s : 0.f;
    float xp = (l < L_ - 1)  ? (x[idx + N_] - m) / s : 0.f;
    float tr = (xm + xc + xp) * (1.f / 3.f);
    trend[idx] = tr; season[idx] = xc - tr;
    spin[(size_t)(b * N_ + n) * L_ + l] = xc;
}

__global__ void mark_idx_kernel(const float* __restrict__ xm,
                                int* __restrict__ tidx, int* __restrict__ didx)
{
    int i = blockIdx.x * blockDim.x + threadIdx.x;
    if (i >= B_ * L_) return;
    tidx[i] = (int)(xm[(size_t)i * 7 + 6] * (float)TODN_);
    didx[i] = (int)(xm[(size_t)i * 7 + 2]);
}

// te/se = [trend|season, tod, dow] @ emb_W + emb_b, laid out [B,N,L,D]
__global__ void embed_kernel(const float* __restrict__ trend, const float* __restrict__ season,
                             const float* __restrict__ xm, const float* __restrict__ embW,
                             const float* __restrict__ embb,
                             float* __restrict__ TE, float* __restrict__ SE)
{
    int t = blockIdx.x; int d = threadIdx.x;
    int l = t % L_; int bn = t / L_; int n = bn % N_; int b = bn / N_;
    size_t mi = (size_t)(b * L_ + l);
    float tod = xm[mi * 7 + 6], dow = xm[mi * 7 + 2];
    size_t si = (size_t)(b * L_ + l) * N_ + n;
    float base = tod * embW[D_ + d] + dow * embW[2 * D_ + d] + embb[d];
    TE[(size_t)t * D_ + d] = trend[si]  * embW[d] + base;
    SE[(size_t)t * D_ + d] = season[si] * embW[d] + base;
}

__global__ void gather_kernel(const int* __restrict__ tidx, const int* __restrict__ didx,
                              const float* __restrict__ tt, const float* __restrict__ dt,
                              float* __restrict__ tg, float* __restrict__ dg)
{
    int i = blockIdx.x; int d = threadIdx.x;
    tg[(size_t)i * D_ + d] = tt[(size_t)tidx[i] * D_ + d];
    dg[(size_t)i * D_ + d] = dt[(size_t)didx[i] * D_ + d];
}

// xe += todP[b,l] + dowP[b,l] + spP[b,n]
__global__ void bcast_add_kernel(float* __restrict__ xe, const float* __restrict__ tp,
                                 const float* __restrict__ dp, const float* __restrict__ spp)
{
    int t = blockIdx.x; int d = threadIdx.x;
    int l = t % L_; int bn = t / L_; int b = bn / N_;
    size_t mi = (size_t)(b * L_ + l);
    xe[(size_t)t * D_ + d] += tp[mi * D_ + d] + dp[mi * D_ + d] + spp[(size_t)bn * D_ + d];
}

// FAVOR+ attention, S=1 (spatial block): o = v * s/(s+1e-6)
__global__ void performer_s1_kernel(const float* __restrict__ Q, const float* __restrict__ Kx,
                                    const float* __restrict__ V, const float* __restrict__ proj,
                                    float* __restrict__ O, int T)
{
    int th = blockIdx.x * blockDim.x + threadIdx.x;
    if (th >= T * H_) return;
    int head = th % H_; int tok = th / H_;
    size_t base = (size_t)tok * D_ + head * DH_;
    float ql[16], kl[16], vl[16], qn = 0.f, kn = 0.f;
    for (int m = 0; m < 16; ++m) {
        ql[m] = Q[base + m] * 0.5f; kl[m] = Kx[base + m] * 0.5f; vl[m] = V[base + m];
        qn += ql[m] * ql[m]; kn += kl[m] * kl[m];
    }
    qn *= 0.5f; kn *= 0.5f;
    float qd[16], kd[16], qmax = -1e30f, kmax = -1e30f;
    for (int j = 0; j < 16; ++j) {
        float a = 0.f, c = 0.f;
        for (int m = 0; m < 16; ++m) { a += ql[m] * proj[m*16+j]; c += kl[m] * proj[m*16+j]; }
        qd[j] = a; kd[j] = c; qmax = fmaxf(qmax, a); kmax = fmaxf(kmax, c);
    }
    float s = 0.f;
    for (int j = 0; j < 16; ++j) {
        float qp = expf(qd[j] - qn - qmax) * 0.25f + 1e-6f;
        float kp = expf(kd[j] - kn - kmax) * 0.25f + 1e-6f;
        s += qp * kp;
    }
    float z = s / (s + 1e-6f);
    for (int d = 0; d < 16; ++d) O[base + d] = vl[d] * z;
}

// FAVOR+ attention, S=96: one block per (seq, head), thread per position l
__global__ void performer_attn_kernel(const float* __restrict__ Q, const float* __restrict__ Kx,
                                      const float* __restrict__ V, const float* __restrict__ proj,
                                      float* __restrict__ O)
{
    int sh = blockIdx.x;
    int head = sh % H_; int seq = sh / H_;
    int l = threadIdx.x;
    bool act = l < L_;
    __shared__ float pr[256];
    __shared__ float kv[256];
    __shared__ float ksum[16];
    __shared__ float red[128];
    for (int i = threadIdx.x; i < 256; i += 128) { pr[i] = proj[i]; kv[i] = 0.f; }
    if (threadIdx.x < 16) ksum[threadIdx.x] = 0.f;
    __syncthreads();

    float ql[16], kl[16], vl[16], qd[16], kd[16], qp[16], kp[16];
    float qn = 0.f, kn = 0.f, qmax = -1e30f, kloc = -1e30f;
    size_t base = 0;
    if (act) {
        base = ((size_t)seq * L_ + l) * D_ + head * DH_;
        for (int m = 0; m < 16; ++m) {
            ql[m] = Q[base + m] * 0.5f; kl[m] = Kx[base + m] * 0.5f; vl[m] = V[base + m];
            qn += ql[m] * ql[m]; kn += kl[m] * kl[m];
        }
        qn *= 0.5f; kn *= 0.5f;
        for (int j = 0; j < 16; ++j) {
            float a = 0.f, c = 0.f;
            for (int m = 0; m < 16; ++m) { a += ql[m] * pr[m*16+j]; c += kl[m] * pr[m*16+j]; }
            qd[j] = a; kd[j] = c;
            qmax = fmaxf(qmax, a); kloc = fmaxf(kloc, c);
        }
    }
    red[threadIdx.x] = act ? kloc : -1e30f;
    __syncthreads();
    for (int s = 64; s > 0; s >>= 1) {
        if (threadIdx.x < s) red[threadIdx.x] = fmaxf(red[threadIdx.x], red[threadIdx.x + s]);
        __syncthreads();
    }
    float kmax = red[0];
    __syncthreads();
    if (act) {
        for (int j = 0; j < 16; ++j) {
            qp[j] = expf(qd[j] - qn - qmax) * 0.25f + 1e-6f;
            kp[j] = expf(kd[j] - kn - kmax) * 0.25f + 1e-6f;
            atomicAdd(&ksum[j], kp[j]);
            for (int d = 0; d < 16; ++d) atomicAdd(&kv[j*16+d], kp[j] * vl[d]);
        }
    }
    __syncthreads();
    if (act) {
        float den = 1e-6f;
        for (int j = 0; j < 16; ++j) den += qp[j] * ksum[j];
        float z = 1.f / den;
        for (int d = 0; d < 16; ++d) {
            float s = 0.f;
            for (int j = 0; j < 16; ++j) s += qp[j] * kv[j*16+d];
            O[base + d] = s * z;
        }
    }
}

// out = LN(X + A) * g + b, one block per row of 128
__global__ void add_ln_kernel(const float* __restrict__ X, const float* __restrict__ A,
                              const float* __restrict__ g, const float* __restrict__ b,
                              float* __restrict__ O)
{
    int t = blockIdx.x; int d = threadIdx.x;
    __shared__ float red[128];
    float v = X[(size_t)t * D_ + d] + A[(size_t)t * D_ + d];
    red[d] = v; __syncthreads();
    for (int s = 64; s > 0; s >>= 1) { if (d < s) red[d] += red[d + s]; __syncthreads(); }
    float mean = red[0] * (1.f / 128.f);
    __syncthreads();
    float df = v - mean;
    red[d] = df * df; __syncthreads();
    for (int s = 64; s > 0; s >>= 1) { if (d < s) red[d] += red[d + s]; __syncthreads(); }
    float var = red[0] * (1.f / 128.f);
    O[(size_t)t * D_ + d] = df * rsqrtf(var + EPS_) * g[d] + b[d];
}

// top-2 softmax gate with renormalized weights -> G[M,4]
__global__ void gate_kernel(const float* __restrict__ X, const float* __restrict__ Wg,
                            float* __restrict__ G, int M)
{
    int t = blockIdx.x * blockDim.x + threadIdx.x;
    if (t >= M) return;
    const float* xr = X + (size_t)t * D_;
    float lg[4] = {0.f, 0.f, 0.f, 0.f};
    for (int d = 0; d < D_; ++d) {
        float xv = xr[d];
        lg[0] += xv * Wg[d*4+0]; lg[1] += xv * Wg[d*4+1];
        lg[2] += xv * Wg[d*4+2]; lg[3] += xv * Wg[d*4+3];
    }
    float mx = fmaxf(fmaxf(lg[0], lg[1]), fmaxf(lg[2], lg[3]));
    float p[4]; float s = 0.f;
    for (int e = 0; e < 4; ++e) { p[e] = expf(lg[e] - mx); s += p[e]; }
    for (int e = 0; e < 4; ++e) p[e] /= s;
    int i0 = 0;
    for (int e = 1; e < 4; ++e) if (p[e] > p[i0]) i0 = e;
    int i1 = -1;
    for (int e = 0; e < 4; ++e) { if (e == i0) continue; if (i1 < 0 || p[e] > p[i1]) i1 = e; }
    float ws = p[i0] + p[i1];
    for (int e = 0; e < 4; ++e)
        G[(size_t)t * 4 + e] = (e == i0) ? p[i0] / ws : (e == i1) ? p[i1] / ws : 0.f;
}

// out[b,p,n] = pre[b,n,p] * std[b] + mean[b]
__global__ void denorm_kernel(const float* __restrict__ pre, const float* __restrict__ mean,
                              const float* __restrict__ stdv, float* __restrict__ out)
{
    int i = blockIdx.x * blockDim.x + threadIdx.x;
    if (i >= SEQ_ * PRED_) return;
    int p = i % PRED_; int bn = i / PRED_; int n = bn % N_; int b = bn / N_;
    out[((size_t)(b * PRED_ + p)) * N_ + n] = pre[i] * stdv[b] + mean[b];
}

// ---------------------------------------------------------------------------
static inline void gemm(hipStream_t st, const float* A, int lda, const float* Bm, int ldb,
                        const float* bias, const float* rs, int rss,
                        float* C, int ldc, int M, int K, int N, int flags)
{
    dim3 g((N + 63) / 64, (M + 63) / 64);
    gemm_ws_kernel<<<g, dim3(256), 0, st>>>(A, lda, Bm, ldb, bias, rs, rss, C, ldc, M, K, N, flags);
}

extern "C" void kernel_launch(void* const* d_in, const int* in_sizes, int n_in,
                              void* d_out, int out_size, void* d_ws, size_t ws_size,
                              hipStream_t stream)
{
    (void)in_sizes; (void)n_in; (void)out_size; (void)ws_size;
    const float* x_enc    = (const float*)d_in[0];
    const float* x_mark   = (const float*)d_in[1];
    const float* emb_W    = (const float*)d_in[2];
    const float* emb_b    = (const float*)d_in[3];
    const float* tod_tab  = (const float*)d_in[4];
    const float* dow_tab  = (const float*)d_in[5];
    const float* adaptive = (const float*)d_in[6];
    const float* sp_W     = (const float*)d_in[7];
    const float* sp_Wq    = (const float*)d_in[8];
    const float* sp_Wk    = (const float*)d_in[9];
    const float* sp_Wv    = (const float*)d_in[10];
    const float* sp_Wo    = (const float*)d_in[11];
    const float* sp_bq    = (const float*)d_in[12];
    const float* sp_bk    = (const float*)d_in[13];
    const float* sp_bv    = (const float*)d_in[14];
    const float* sp_bo    = (const float*)d_in[15];
    const float* sp_proj  = (const float*)d_in[16];
    const float* sp_gate  = (const float*)d_in[17];
    const float* sp_W1    = (const float*)d_in[18];
    const float* sp_b1    = (const float*)d_in[19];
    const float* sp_W2    = (const float*)d_in[20];
    const float* sp_b2    = (const float*)d_in[21];
    const float* sp_ln2g  = (const float*)d_in[22];
    const float* sp_ln2b  = (const float*)d_in[23];
    const float* sp_ln3g  = (const float*)d_in[24];
    const float* sp_ln3b  = (const float*)d_in[25];
    const float* proj_W   = (const float*)d_in[26];
    const float* proj_b   = (const float*)d_in[27];
    const float* LWq      = (const float*)d_in[28];
    const float* LWk      = (const float*)d_in[29];
    const float* LWv      = (const float*)d_in[30];
    const float* LWo      = (const float*)d_in[31];
    const float* Lbq      = (const float*)d_in[32];
    const float* Lbk      = (const float*)d_in[33];
    const float* Lbv      = (const float*)d_in[34];
    const float* Lbo      = (const float*)d_in[35];
    const float* Lproj    = (const float*)d_in[36];
    const float* Lgate    = (const float*)d_in[37];
    const float* LW1      = (const float*)d_in[38];
    const float* Lb1      = (const float*)d_in[39];
    const float* LW2      = (const float*)d_in[40];
    const float* Lb2      = (const float*)d_in[41];
    const float* Lln2g    = (const float*)d_in[42];
    const float* Lln2b    = (const float*)d_in[43];
    const float* Lln3g    = (const float*)d_in[44];
    const float* Lln3b    = (const float*)d_in[45];
    const float* mix_W    = (const float*)d_in[46];
    const float* mix_b    = (const float*)d_in[47];

    // workspace layout (floats)
    float* W = (float*)d_ws;
    size_t o = 0;
    float* w_mean  = W + o; o += 4;
    float* w_std   = W + o; o += 4;
    float* w_trend = W + o; o += BLN_;
    float* w_seas  = W + o; o += BLN_;
    float* w_spin  = W + o; o += BLN_;
    int*   w_tidx  = (int*)(W + o); o += B_ * L_;
    int*   w_didx  = (int*)(W + o); o += B_ * L_;
    float* w_tg    = W + o; o += (size_t)B_ * L_ * D_;
    float* w_dg    = W + o; o += (size_t)B_ * L_ * D_;
    float* w_tp    = W + o; o += (size_t)B_ * L_ * D_;
    float* w_dp    = W + o; o += (size_t)B_ * L_ * D_;
    float* w_sp0   = W + o; o += (size_t)SEQ_ * D_;
    float* w_spq   = W + o; o += (size_t)SEQ_ * D_;
    float* w_spk   = W + o; o += (size_t)SEQ_ * D_;
    float* w_spv   = W + o; o += (size_t)SEQ_ * D_;
    float* w_spt1  = W + o; o += (size_t)SEQ_ * D_;
    float* w_spt2  = W + o; o += (size_t)SEQ_ * D_;
    float* w_gsp   = W + o; o += (size_t)SEQ_ * E_;
    float* w_g     = W + o; o += (size_t)TOK_ * E_;
    float* BX      = W + o; o += (size_t)TOK_ * D_;   // running activation x
    float* Bq      = W + o; o += (size_t)TOK_ * D_;   // q / te / attn-pre-LN
    float* Bk      = W + o; o += (size_t)TOK_ * D_;   // k / se / MoE h-chunk
    float* Bv      = W + o; o += (size_t)TOK_ * D_;   // v / MoE accum
    float* Bo4     = W + o; o += (size_t)TOK_ * D_;   // attention output / mixer out

    // ---- RevIN, decomposition, marks ----
    revin_stats_kernel<<<B_, 256, 0, stream>>>(x_enc, w_mean, w_std);
    prep_kernel<<<(BLN_ + 255) / 256, 256, 0, stream>>>(x_enc, w_mean, w_std,
                                                        w_trend, w_seas, w_spin);
    mark_idx_kernel<<<(B_ * L_ + 255) / 256, 256, 0, stream>>>(x_mark, w_tidx, w_didx);
    embed_kernel<<<TOK_, D_, 0, stream>>>(w_trend, w_seas, x_mark, emb_W, emb_b, Bq, Bk);
    gather_kernel<<<B_ * L_, D_, 0, stream>>>(w_tidx, w_didx, tod_tab, dow_tab, w_tg, w_dg);

    // ---- spatial transformer block (S = 1) ----
    gemm(stream, w_spin, L_, sp_W, D_, nullptr, nullptr, 0, w_sp0, D_, SEQ_, L_, D_, 0);
    gemm(stream, w_sp0, D_, sp_Wq, D_, sp_bq, nullptr, 0, w_spq, D_, SEQ_, D_, D_, 0);
    gemm(stream, w_sp0, D_, sp_Wk, D_, sp_bk, nullptr, 0, w_spk, D_, SEQ_, D_, D_, 0);
    gemm(stream, w_sp0, D_, sp_Wv, D_, sp_bv, nullptr, 0, w_spv, D_, SEQ_, D_, D_, 0);
    performer_s1_kernel<<<(SEQ_ * H_ + 255) / 256, 256, 0, stream>>>(w_spq, w_spk, w_spv,
                                                                     sp_proj, w_spt1, SEQ_);
    gemm(stream, w_spt1, D_, sp_Wo, D_, sp_bo, nullptr, 0, w_spt2, D_, SEQ_, D_, D_, 0);
    add_ln_kernel<<<SEQ_, D_, 0, stream>>>(w_sp0, w_spt2, sp_ln2g, sp_ln2b, w_spq);
    gate_kernel<<<(SEQ_ + 255) / 256, 256, 0, stream>>>(w_spq, sp_gate, w_gsp, SEQ_);
    for (int e = 0; e < E_; ++e) {
        for (int c = 0; c < DFF_ / 128; ++c) {
            gemm(stream, w_spq, D_,
                 sp_W1 + (size_t)e * D_ * DFF_ + c * 128, DFF_,
                 sp_b1 + (size_t)e * DFF_ + c * 128, nullptr, 0,
                 w_spt1, 128, SEQ_, D_, 128, GF_RELU);
            gemm(stream, w_spt1, 128,
                 sp_W2 + (size_t)e * DFF_ * D_ + (size_t)c * 128 * D_, D_,
                 (c == 0) ? (sp_b2 + (size_t)e * D_) : nullptr, w_gsp + e, E_,
                 w_spt2, D_, SEQ_, 128, D_, (e == 0 && c == 0) ? 0 : GF_ACC);
        }
    }
    add_ln_kernel<<<SEQ_, D_, 0, stream>>>(w_spq, w_spt2, sp_ln3g, sp_ln3b, w_sp0);

    // ---- embedding projection xe = concat6 @ proj_W + proj_b ----
    gemm(stream, w_tg, D_, proj_W + (size_t)3 * D_ * D_, D_, nullptr, nullptr, 0,
         w_tp, D_, B_ * L_, D_, D_, 0);
    gemm(stream, w_dg, D_, proj_W + (size_t)4 * D_ * D_, D_, nullptr, nullptr, 0,
         w_dp, D_, B_ * L_, D_, D_, 0);
    gemm(stream, w_sp0, D_, proj_W + (size_t)5 * D_ * D_, D_, nullptr, nullptr, 0,
         w_spt1, D_, SEQ_, D_, D_, 0);
    gemm(stream, Bq /*te*/, D_, proj_W, D_, proj_b, nullptr, 0, BX, D_, TOK_, D_, D_, 0);
    gemm(stream, Bk /*se*/, D_, proj_W + (size_t)1 * D_ * D_, D_, nullptr, nullptr, 0,
         BX, D_, TOK_, D_, D_, GF_ACC);
    for (int b = 0; b < B_; ++b)
        gemm(stream, adaptive, D_, proj_W + (size_t)2 * D_ * D_, D_, nullptr, nullptr, 0,
             BX + (size_t)b * N_ * L_ * D_, D_, N_ * L_, D_, D_, GF_ACC);
    bcast_add_kernel<<<TOK_, D_, 0, stream>>>(BX, w_tp, w_dp, w_spt1);

    // ---- main transformer layers ----
    for (int i = 0; i < LAYERS_; ++i) {
        const float* Wq = LWq + (size_t)i * D_ * D_;
        const float* Wk = LWk + (size_t)i * D_ * D_;
        const float* Wv = LWv + (size_t)i * D_ * D_;
        const float* Wo = LWo + (size_t)i * D_ * D_;
        gemm(stream, BX, D_, Wq, D_, Lbq + (size_t)i * D_, nullptr, 0, Bq, D_, TOK_, D_, D_, 0);
        gemm(stream, BX, D_, Wk, D_, Lbk + (size_t)i * D_, nullptr, 0, Bk, D_, TOK_, D_, D_, 0);
        gemm(stream, BX, D_, Wv, D_, Lbv + (size_t)i * D_, nullptr, 0, Bv, D_, TOK_, D_, D_, 0);
        performer_attn_kernel<<<SEQ_ * H_, 128, 0, stream>>>(Bq, Bk, Bv,
                                                             Lproj + (size_t)i * 256, Bo4);
        gemm(stream, Bo4, D_, Wo, D_, Lbo + (size_t)i * D_, nullptr, 0, Bq, D_, TOK_, D_, D_, 0);
        add_ln_kernel<<<TOK_, D_, 0, stream>>>(BX, Bq, Lln2g + (size_t)i * D_,
                                               Lln2b + (size_t)i * D_, BX);
        gate_kernel<<<(TOK_ + 255) / 256, 256, 0, stream>>>(BX, Lgate + (size_t)i * D_ * E_,
                                                            w_g, TOK_);
        for (int e = 0; e < E_; ++e) {
            for (int c = 0; c < DFF_ / 128; ++c) {
                gemm(stream, BX, D_,
                     LW1 + ((size_t)i * E_ + e) * D_ * DFF_ + c * 128, DFF_,
                     Lb1 + ((size_t)i * E_ + e) * DFF_ + c * 128, nullptr, 0,
                     Bk, 128, TOK_, D_, 128, GF_RELU);
                gemm(stream, Bk, 128,
                     LW2 + ((size_t)i * E_ + e) * DFF_ * D_ + (size_t)c * 128 * D_, D_,
                     (c == 0) ? (Lb2 + ((size_t)i * E_ + e) * D_) : nullptr, w_g + e, E_,
                     Bv, D_, TOK_, 128, D_, (e == 0 && c == 0) ? 0 : GF_ACC);
            }
        }
        add_ln_kernel<<<TOK_, D_, 0, stream>>>(BX, Bv, Lln3g + (size_t)i * D_,
                                               Lln3b + (size_t)i * D_, BX);
    }

    // ---- output mixer + RevIN denorm ----
    gemm(stream, BX, L_ * D_, mix_W, PRED_, mix_b, nullptr, 0,
         Bo4, PRED_, SEQ_, L_ * D_, PRED_, 0);
    denorm_kernel<<<(SEQ_ * PRED_ + 255) / 256, 256, 0, stream>>>(Bo4, w_mean, w_std,
                                                                  (float*)d_out);
}
